// MaskedMultiSelfAttention_32865089749259
// MI455X (gfx1250) — matrix operand order repaired and checked by reading, not compiled, not run on hardware
//
#include <hip/hip_runtime.h>
#include <hip/hip_bf16.h>

typedef __attribute__((ext_vector_type(16))) _Float16 v16h;
typedef __attribute__((ext_vector_type(8)))  _Float16 h8;
typedef __attribute__((ext_vector_type(8)))  float    v8f;

#define BATCH   4
#define TSEQ    2048
#define DMODEL  1024
#define NHEAD   16
#define HD      64
#define ATT_SCALE 8.0f   /* faithful to source bug: multiply by sqrt(head_dim) */

// ---------------------------------------------------------------------------
// Stage 1: QKV projection.  out = x @ W^T + b, cast to f16.
//   z==0 -> Q  [B,H,T,HD]
//   z==1 -> K  [B,H,T,HD]
//   z==2 -> V^T[B,H,HD,T]   (transposed so PV B-fragments load contiguously)
// One wave computes a 64x64 output block: 4x4 accumulator tiles, 4 A-frags x
// 4 B-frags per K=32 step -> 16 v_wmma per 8 fragment loads (4x operand reuse).
// ---------------------------------------------------------------------------
__global__ __launch_bounds__(32)
void qkv_proj_kernel(const float* __restrict__ x,
                     const float* __restrict__ Wq, const float* __restrict__ bq,
                     const float* __restrict__ Wk, const float* __restrict__ bk,
                     const float* __restrict__ Wv, const float* __restrict__ bv,
                     _Float16* __restrict__ Qh,
                     _Float16* __restrict__ Kh,
                     _Float16* __restrict__ Vt)
{
    const int lane = threadIdx.x;
    const int lo = lane & 15;
    const int hi = lane >> 4;

    const int nb = blockIdx.x;   // 0..15  : 64-wide output-feature block
    const int mb = blockIdx.y;   // 0..127 : 64-tall row block over B*T
    const int z  = blockIdx.z;   // 0:Q 1:K 2:V

    const float* W    = (z == 0) ? Wq : (z == 1) ? Wk : Wv;
    const float* bias = (z == 0) ? bq : (z == 1) ? bk : bv;
    _Float16*    out  = (z == 0) ? Qh : (z == 1) ? Kh : Vt;

    // Per-lane source rows for the 4 A subtiles (rows of x) and 4 B subtiles
    // (rows of W supply columns of B: B[k][n] = W[n][k]).
    const float* __restrict__ arow[4];
    const float* __restrict__ brow[4];
    #pragma unroll
    for (int i = 0; i < 4; ++i) {
        arow[i] = x + (size_t)(mb * 64 + i * 16 + lo) * DMODEL;
        brow[i] = W + (size_t)(nb * 64 + i * 16 + lo) * DMODEL;
    }

    v8f acc[4][4] = {};
    for (int kt = 0; kt < DMODEL; kt += 32) {
        v16h a[4], b[4];
        #pragma unroll
        for (int i = 0; i < 4; ++i) {
            // A fragment: lanes 0-15 hold K=0..15, lanes 16-31 hold K=16..31
            const float* ga = arow[i] + kt + hi * 16;
            const float* gb = arow[i] + kt + hi * 16 + 8;
            #pragma unroll
            for (int e = 0; e < 8; ++e) {
                a[i][e]     = (_Float16)ga[e];
                a[i][e + 8] = (_Float16)gb[e];
            }
            // B fragment: lanes 0-15 hold K=0..15, lanes 16-31 hold K=16..31
            const float* gw = brow[i] + kt + hi * 16;
            #pragma unroll
            for (int e = 0; e < 16; ++e) b[i][e] = (_Float16)gw[e];
        }
        #pragma unroll
        for (int i = 0; i < 4; ++i)
            #pragma unroll
            for (int j = 0; j < 4; ++j)
                acc[i][j] = __builtin_amdgcn_wmma_f32_16x16x32_f16(
                    false, a[i], false, b[j], (short)0, acc[i][j], false, false);
    }

    // Epilogue: C/D layout element (vgpr r, lane) -> row = r + 8*hi, col = lo
    #pragma unroll
    for (int j = 0; j < 4; ++j) {
        const int nf = nb * 64 + j * 16 + lo;     // global output feature
        const float bn = bias[nf];
        const int h = nf >> 6;                    // head
        const int d = nf & (HD - 1);              // within-head dim
        #pragma unroll
        for (int i = 0; i < 4; ++i) {
            #pragma unroll
            for (int r = 0; r < 8; ++r) {
                const int mrow = mb * 64 + i * 16 + r + 8 * hi;  // b*T + t
                const int bb = mrow >> 11;
                const int t  = mrow & (TSEQ - 1);
                const _Float16 val = (_Float16)(acc[i][j][r] + bn);
                size_t off;
                if (z < 2) off = (((size_t)(bb * NHEAD + h)) * TSEQ + t) * HD + d;
                else       off = (((size_t)(bb * NHEAD + h)) * HD + d) * TSEQ + t;
                out[off] = val;
            }
        }
    }
}

// ---------------------------------------------------------------------------
// Stage 2: causal flash attention.  One wave per 16-query tile per (b,h).
// Per 32-key step: 4 WMMA for QK^T (K=64 as 2x32, two 16-key tiles),
// scale*8, causal mask, online softmax (16-lane shfl_xor reductions),
// P transposed through LDS into A-fragment layout, 4 WMMA for P@V.
// ---------------------------------------------------------------------------
__global__ __launch_bounds__(32)
void attn_kernel(const _Float16* __restrict__ Qh,
                 const _Float16* __restrict__ Kh,
                 const _Float16* __restrict__ Vt,
                 float* __restrict__ out)
{
    __shared__ _Float16 pbuf[16 * 32];   // P tile, row-major [16 q rows][32 keys]

    const int lane = threadIdx.x;
    const int lo = lane & 15;
    const int hi = lane >> 4;

    const int qt = blockIdx.x;           // 0..127 query tile
    const int bh = blockIdx.y;           // 0..63  b*NHEAD + h
    const int b  = bh >> 4;
    const int h  = bh & 15;
    const int q0 = qt * 16;

    const _Float16* __restrict__ qbase = Qh + (size_t)bh * TSEQ * HD;
    const _Float16* __restrict__ kbase = Kh + (size_t)bh * TSEQ * HD;
    const _Float16* __restrict__ vbase = Vt + (size_t)bh * HD * TSEQ;

    // Q A-fragments (16 x 64 over two K=32 chunks), resident for the tile.
    v16h qa[2];
    {
        const _Float16* qrow = qbase + (size_t)(q0 + lo) * HD;
        #pragma unroll
        for (int f = 0; f < 2; ++f) {
            h8 g0 = *(const h8*)(qrow + f * 32 + hi * 16);
            h8 g1 = *(const h8*)(qrow + f * 32 + hi * 16 + 8);
            #pragma unroll
            for (int e = 0; e < 8; ++e) { qa[f][e] = g0[e]; qa[f][e + 8] = g1[e]; }
        }
    }

    v8f acc[4] = {};                     // O tile: 4 hd n-tiles of 16
    float mi[8], li[8];
    #pragma unroll
    for (int r = 0; r < 8; ++r) { mi[r] = -1e30f; li[r] = 0.0f; }

    const int q_hi   = q0 + 15;
    const int nsteps = (q0 + 16 + 31) / 32;

    for (int j = 0; j < nsteps * 32; j += 32) {
        const bool do_s1 = (j + 16) <= q_hi;   // uniform across wave

        // ---- S = Q K^T for keys [j, j+31] ----
        v8f s0 = {}, s1 = {};
        {
            const _Float16* krow0 = kbase + (size_t)(j + lo) * HD;
            #pragma unroll
            for (int c = 0; c < 2; ++c) {
                v16h kb;
                h8 g0 = *(const h8*)(krow0 + c * 32 + hi * 16);
                h8 g1 = *(const h8*)(krow0 + c * 32 + hi * 16 + 8);
                #pragma unroll
                for (int e = 0; e < 8; ++e) { kb[e] = g0[e]; kb[e + 8] = g1[e]; }
                s0 = __builtin_amdgcn_wmma_f32_16x16x32_f16(false, qa[c], false, kb,
                                                            (short)0, s0, false, false);
            }
            if (do_s1) {
                const _Float16* krow1 = kbase + (size_t)(j + 16 + lo) * HD;
                #pragma unroll
                for (int c = 0; c < 2; ++c) {
                    v16h kb;
                    h8 g0 = *(const h8*)(krow1 + c * 32 + hi * 16);
                    h8 g1 = *(const h8*)(krow1 + c * 32 + hi * 16 + 8);
                    #pragma unroll
                    for (int e = 0; e < 8; ++e) { kb[e] = g0[e]; kb[e + 8] = g1[e]; }
                    s1 = __builtin_amdgcn_wmma_f32_16x16x32_f16(false, qa[c], false, kb,
                                                                (short)0, s1, false, false);
                }
            }
        }

        // ---- scale, causal mask, per-row max over this 32-key step ----
        float pm[8];
        #pragma unroll
        for (int r = 0; r < 8; ++r) {
            const int q  = q0 + r + 8 * hi;
            const int k0 = j + lo;
            const int k1 = j + 16 + lo;
            float a0 = s0[r] * ATT_SCALE;
            if (k0 > q) a0 = -1e30f;
            float a1 = do_s1 ? s1[r] * ATT_SCALE : -1e30f;
            if (k1 > q) a1 = -1e30f;
            s0[r] = a0; s1[r] = a1;
            pm[r] = fmaxf(a0, a1);
        }
        #pragma unroll
        for (int off = 1; off < 16; off <<= 1) {
            #pragma unroll
            for (int r = 0; r < 8; ++r)
                pm[r] = fmaxf(pm[r], __shfl_xor(pm[r], off, 32));
        }

        // ---- online softmax update; write P into LDS for transpose ----
        float alpha[8], ps[8];
        #pragma unroll
        for (int r = 0; r < 8; ++r) {
            const float mnew = fmaxf(mi[r], pm[r]);
            alpha[r] = __expf(mi[r] - mnew);
            mi[r] = mnew;
            const float p0 = __expf(s0[r] - mnew);
            const float p1 = do_s1 ? __expf(s1[r] - mnew) : 0.0f;
            const int row = r + 8 * hi;
            pbuf[row * 32 + lo]      = (_Float16)p0;
            pbuf[row * 32 + 16 + lo] = (_Float16)p1;
            ps[r] = p0 + p1;
        }
        #pragma unroll
        for (int off = 1; off < 16; off <<= 1) {
            #pragma unroll
            for (int r = 0; r < 8; ++r)
                ps[r] += __shfl_xor(ps[r], off, 32);
        }
        #pragma unroll
        for (int r = 0; r < 8; ++r) li[r] = li[r] * alpha[r] + ps[r];

        __syncthreads();
        // P as A-fragment (16x32): lane row = lo, contiguous K = 16*hi + e
        v16h pa;
        #pragma unroll
        for (int e = 0; e < 8; ++e) {
            pa[e]     = pbuf[lo * 32 + hi * 16 + e];
            pa[e + 8] = pbuf[lo * 32 + hi * 16 + 8 + e];
        }
        __syncthreads();

        // ---- O += P @ V (V^T layout gives contiguous B-fragments) ----
        #pragma unroll
        for (int n2 = 0; n2 < 4; ++n2) {
            const _Float16* vrow = vbase + (size_t)(n2 * 16 + lo) * TSEQ + j;
            v16h vb;
            h8 g0 = *(const h8*)(vrow + hi * 16);
            h8 g1 = *(const h8*)(vrow + hi * 16 + 8);
            #pragma unroll
            for (int e = 0; e < 8; ++e) { vb[e] = g0[e]; vb[e + 8] = g1[e]; }
            v8f cin;
            #pragma unroll
            for (int r = 0; r < 8; ++r) cin[r] = acc[n2][r] * alpha[r];
            acc[n2] = __builtin_amdgcn_wmma_f32_16x16x32_f16(false, pa, false, vb,
                                                             (short)0, cin, false, false);
        }
    }

    // ---- epilogue: divide by softmax sum, write fp32 [B,T,D] ----
    #pragma unroll
    for (int r = 0; r < 8; ++r) {
        const float inv = 1.0f / li[r];
        const int q = q0 + r + 8 * hi;
        const size_t obase = ((size_t)b * TSEQ + q) * DMODEL + h * HD;
        #pragma unroll
        for (int n2 = 0; n2 < 4; ++n2)
            out[obase + n2 * 16 + lo] = acc[n2][r] * inv;
    }
}

// ---------------------------------------------------------------------------
extern "C" void kernel_launch(void* const* d_in, const int* in_sizes, int n_in,
                              void* d_out, int out_size, void* d_ws, size_t ws_size,
                              hipStream_t stream)
{
    const float* x  = (const float*)d_in[0];
    const float* Wq = (const float*)d_in[1];
    const float* bq = (const float*)d_in[2];
    const float* Wk = (const float*)d_in[3];
    const float* bk = (const float*)d_in[4];
    const float* Wv = (const float*)d_in[5];
    const float* bv = (const float*)d_in[6];
    float* out = (float*)d_out;

    const size_t elems = (size_t)BATCH * NHEAD * TSEQ * HD;   // 8,388,608
    _Float16* Qh = (_Float16*)d_ws;
    _Float16* Kh = Qh + elems;
    _Float16* Vt = Kh + elems;   // 48 MB total of f16 scratch

    dim3 grid1(DMODEL / 64, (BATCH * TSEQ) / 64, 3);   // 16 x 128 x 3 blocks
    qkv_proj_kernel<<<grid1, 32, 0, stream>>>(x, Wq, bq, Wk, bk, Wv, bv, Qh, Kh, Vt);

    dim3 grid2(TSEQ / 16, BATCH * NHEAD);              // 128 x 64
    attn_kernel<<<grid2, 32, 0, stream>>>(Qh, Kh, Vt, out);
}
